// DeepSetsEnsemble_52793738002550
// MI455X (gfx1250) — matrix operand order repaired and checked by reading, not compiled, not run on hardware
//
#include <hip/hip_runtime.h>

// ---------------- problem constants ----------------
#define B_   8192
#define PTS_ 128
#define T_   (B_ * PTS_)      // 1048576
#define TPW  8                // tiles (of 16 points) per wave in phi kernel

typedef __attribute__((ext_vector_type(16))) _Float16 v16h;
typedef __attribute__((ext_vector_type(8)))  float    v8f;

#define WMMA_F16(a, b, c) \
  __builtin_amdgcn_wmma_f32_16x16x32_f16(false, (a), false, (b), (short)0, (c), false, false)

// A-matrix K index for (dword vg, lane-half hi): lanes 0-15 hold K 0..7,16..23
__device__ __forceinline__ int kA(int vg, int hi) {
  return (vg < 4 ? 2 * vg : 16 + 2 * (vg - 4)) + 8 * hi;
}
// B-matrix K index: same as A: lanes 0-15 hold K 0..7,16..23
__device__ __forceinline__ int kB(int vg, int hi) {
  return (vg < 4 ? 2 * vg : 16 + 2 * (vg - 4)) + 8 * hi;
}

// ---------------- utility kernels ----------------
__global__ void zero_kernel(float* __restrict__ p, int n) {
  int i = blockIdx.x * blockDim.x + threadIdx.x;
  if (i < n) p[i] = 0.f;
}

// exclusive prefix sum of 8192 lengths, single block of 256 threads
__global__ __launch_bounds__(256) void scan_offsets(const int* __restrict__ lengths,
                                                    int* __restrict__ offsets) {
  __shared__ int chunk[256];
  const int tid = threadIdx.x;
  const int base = tid * 32;
  int local[32];
  int s = 0;
  for (int i = 0; i < 32; ++i) { local[i] = s; s += lengths[base + i]; }
  chunk[tid] = s;
  __syncthreads();
  for (int off = 1; off < 256; off <<= 1) {
    int v = chunk[tid];
    int add = (tid >= off) ? chunk[tid - off] : 0;
    __syncthreads();
    chunk[tid] = v + add;
    __syncthreads();
  }
  const int cbase = (tid == 0) ? 0 : chunk[tid - 1];
  for (int i = 0; i < 32; ++i) offsets[base + i] = cbase + local[i];
  if (tid == 255) offsets[B_] = chunk[255];
}

__global__ void seg_fill(const int* __restrict__ offsets, int* __restrict__ seg) {
  int b = blockIdx.x * blockDim.x + threadIdx.x;
  if (b >= B_) return;
  int s = offsets[b], e = offsets[b + 1];
  for (int t = s; t < e; ++t) seg[t] = b;
}

// ---------------- fused phi-MLP + ragged segment-sum ----------------
// block = 256 threads = 8 waves; each wave processes TPW consecutive 16-point tiles.
__global__ __launch_bounds__(256) void phi_pool_kernel(
    const float* __restrict__ x,      // [16, T] channel-major
    const float* __restrict__ w1,     // [64,16]
    const float* __restrict__ b1,     // [64]
    const float* __restrict__ w2,     // [64,64]
    const float* __restrict__ b2,     // [64]
    const int*   __restrict__ seg,    // [T]
    float* __restrict__ pooled)       // [B,64] (atomic accumulate)
{
  __shared__ _Float16 sW1[64 * 16];
  __shared__ _Float16 sW2[64 * 64];
  __shared__ float sB1[64], sB2[64];
  __shared__ _Float16 sH1[8][16 * 64];   // per-wave 16x64 f16 staging

  const int tid = threadIdx.x;
  for (int i = tid; i < 64 * 16; i += 256) sW1[i] = (_Float16)w1[i];
  for (int i = tid; i < 64 * 64; i += 256) sW2[i] = (_Float16)w2[i];
  if (tid < 64) { sB1[tid] = b1[tid]; sB2[tid] = b2[tid]; }
  __syncthreads();

  const int wave = tid >> 5;
  const int lane = tid & 31;
  const int lm   = lane & 15;
  const int hi   = lane >> 4;

  const v8f zero8 = {0.f, 0.f, 0.f, 0.f, 0.f, 0.f, 0.f, 0.f};

  // hoisted B fragments: phi1 (4 n-tiles, K=32 with upper half zero-padded)
  v16h bf1[4];
  // phi2 (4 n-tiles x 2 K-chunks of 32)
  v16h bf2[4][2];
#pragma unroll
  for (int nt = 0; nt < 4; ++nt) {
    const int row = nt * 16 + lm;   // output channel (B-matrix N)
#pragma unroll
    for (int vg = 0; vg < 8; ++vg) {
      const int k = kB(vg, hi);
      _Float16 lo = (_Float16)0.f, hh = (_Float16)0.f;
      if (k < 16) { lo = sW1[row * 16 + k]; hh = sW1[row * 16 + k + 1]; }
      bf1[nt][2 * vg]     = lo;
      bf1[nt][2 * vg + 1] = hh;
    }
#pragma unroll
    for (int kc = 0; kc < 2; ++kc) {
#pragma unroll
      for (int vg = 0; vg < 8; ++vg) {
        const int k = kc * 32 + kB(vg, hi);
        bf2[nt][kc][2 * vg]     = sW2[row * 64 + k];
        bf2[nt][kc][2 * vg + 1] = sW2[row * 64 + k + 1];
      }
    }
  }

  _Float16* myH = sH1[wave];

  // run-length merged pooling state (per lane: 4 channels, one seg id)
  int   curSeg = -1;
  float acc[4] = {0.f, 0.f, 0.f, 0.f};

  const long long baseTile = ((long long)blockIdx.x * 8 + wave) * TPW;
#pragma unroll 1
  for (int it = 0; it < TPW; ++it) {
    const int t0 = (int)((baseTile + it) * 16);

    if (it + 1 < TPW)   // stream-ahead hint -> global_prefetch_b8
      __builtin_prefetch(x + (size_t)lm * T_ + t0 + 16, 0, 3);

    // ---- A fragment for phi1: 16 points x K=32 (channels 0..15, padded) ----
    v16h a1;
#pragma unroll
    for (int vg = 0; vg < 8; ++vg) {
      const int k = kA(vg, hi);
      float lo = 0.f, hh = 0.f;
      if (k < 16) {
        lo = x[(size_t)k * T_ + t0 + lm];
        hh = x[(size_t)(k + 1) * T_ + t0 + lm];
      }
      a1[2 * vg]     = (_Float16)lo;
      a1[2 * vg + 1] = (_Float16)hh;
    }

    // ---- phi1: 4 WMMAs, bias+ReLU, stage f16 result in LDS ----
#pragma unroll
    for (int nt = 0; nt < 4; ++nt) {
      v8f d = zero8;
      d = WMMA_F16(a1, bf1[nt], d);
      const int col = nt * 16 + lm;
      const float bias = sB1[col];
#pragma unroll
      for (int j = 0; j < 8; ++j) {
        float v = d[j] + bias;
        v = v > 0.f ? v : 0.f;
        myH[(8 * hi + j) * 64 + col] = (_Float16)v;
      }
    }

    // ---- phi2: reload A frags from LDS (intra-wave DS ordering), 8 WMMAs ----
    v8f d2[4];
#pragma unroll
    for (int nt = 0; nt < 4; ++nt) d2[nt] = zero8;
#pragma unroll
    for (int kc = 0; kc < 2; ++kc) {
      v16h a2;
#pragma unroll
      for (int vg = 0; vg < 8; ++vg) {
        const int k = kc * 32 + kA(vg, hi);
        a2[2 * vg]     = myH[lm * 64 + k];
        a2[2 * vg + 1] = myH[lm * 64 + k + 1];
      }
#pragma unroll
      for (int nt = 0; nt < 4; ++nt) d2[nt] = WMMA_F16(a2, bf2[nt][kc], d2[nt]);
    }

    // ---- ragged pool: run-length merge over this lane's 8 rows ----
#pragma unroll
    for (int j = 0; j < 8; ++j) {
      const int p = t0 + 8 * hi + j;
      const int s = seg[p];
      float v[4];
#pragma unroll
      for (int nt = 0; nt < 4; ++nt) {
        float u = d2[nt][j] + sB2[nt * 16 + lm];
        v[nt] = u > 0.f ? u : 0.f;
      }
      if (s != curSeg) {
        if (curSeg >= 0) {
#pragma unroll
          for (int nt = 0; nt < 4; ++nt)
            atomicAdd(&pooled[(size_t)curSeg * 64 + nt * 16 + lm], acc[nt]);
        }
        curSeg = s;
#pragma unroll
        for (int nt = 0; nt < 4; ++nt) acc[nt] = v[nt];
      } else {
#pragma unroll
        for (int nt = 0; nt < 4; ++nt) acc[nt] += v[nt];
      }
    }
  }
  if (curSeg >= 0) {
#pragma unroll
    for (int nt = 0; nt < 4; ++nt)
      atomicAdd(&pooled[(size_t)curSeg * 64 + nt * 16 + lm], acc[nt]);
  }
}

// ---------------- mean-pool finalize + scalar concat -> f16 z [B,96] ----------------
__global__ void zfin_kernel(const float* __restrict__ pooled,
                            const float* __restrict__ x_scalar,
                            const int*   __restrict__ lengths,
                            _Float16* __restrict__ zh) {
  int i = blockIdx.x * blockDim.x + threadIdx.x;
  if (i >= B_ * 96) return;
  int b = i / 96, c = i - b * 96;
  float v = 0.f;
  if (c < 64)      v = pooled[(size_t)b * 64 + c] / (float)lengths[b];
  else if (c < 72) v = x_scalar[(size_t)b * 8 + (c - 64)];
  zh[i] = (_Float16)v;
}

// ---------------- head layer 1: [B,72(->96)] x [96,128] + col stats ----------------
__global__ __launch_bounds__(256) void head1_kernel(
    const _Float16* __restrict__ zh,   // [B,96]
    const float* __restrict__ w1,      // [128,72]
    const float* __restrict__ b1,      // [128]
    float* __restrict__ u1,            // [B,128]
    float* __restrict__ stat1)         // [256]: sum | sumsq
{
  __shared__ _Float16 sW[128 * 96];
  const int tid = threadIdx.x;
  for (int i = tid; i < 128 * 96; i += 256) {
    int r = i / 96, c = i - r * 96;
    sW[i] = (_Float16)(c < 72 ? w1[r * 72 + c] : 0.f);
  }
  __syncthreads();

  const int wave = tid >> 5, lane = tid & 31, lm = lane & 15, hi = lane >> 4;
  const int r0 = (blockIdx.x * 8 + wave) * 16;
  const v8f zero8 = {0.f, 0.f, 0.f, 0.f, 0.f, 0.f, 0.f, 0.f};

  v16h af[3];
#pragma unroll
  for (int kc = 0; kc < 3; ++kc) {
#pragma unroll
    for (int vg = 0; vg < 8; ++vg) {
      const int k = kc * 32 + kA(vg, hi);
      af[kc][2 * vg]     = zh[(size_t)(r0 + lm) * 96 + k];
      af[kc][2 * vg + 1] = zh[(size_t)(r0 + lm) * 96 + k + 1];
    }
  }
#pragma unroll
  for (int nt = 0; nt < 8; ++nt) {
    v8f d = zero8;
#pragma unroll
    for (int kc = 0; kc < 3; ++kc) {
      v16h bf;
#pragma unroll
      for (int vg = 0; vg < 8; ++vg) {
        const int k = kc * 32 + kB(vg, hi);
        bf[2 * vg]     = sW[(nt * 16 + lm) * 96 + k];
        bf[2 * vg + 1] = sW[(nt * 16 + lm) * 96 + k + 1];
      }
      d = WMMA_F16(af[kc], bf, d);
    }
    const int c = nt * 16 + lm;
    const float bias = b1[c];
    float lsum = 0.f, lsq = 0.f;
#pragma unroll
    for (int j = 0; j < 8; ++j) {
      float v = d[j] + bias;
      u1[(size_t)(r0 + 8 * hi + j) * 128 + c] = v;
      lsum += v; lsq += v * v;
    }
    atomicAdd(&stat1[c], lsum);
    atomicAdd(&stat1[128 + c], lsq);
  }
}

// ---------------- batchnorm finalize: stats -> (scale, shift) ----------------
__global__ void bnfin_kernel(const float* __restrict__ stat,
                             const float* __restrict__ g,
                             const float* __restrict__ bb,
                             float* __restrict__ bn, int C, float Bf) {
  int c = threadIdx.x;
  if (c >= C) return;
  float mean = stat[c] / Bf;
  float var  = stat[C + c] / Bf - mean * mean;
  float rstd = rsqrtf(var + 1e-5f);
  float sc = g[c] * rstd;
  bn[c]     = sc;
  bn[C + c] = bb[c] - mean * sc;
}

// ---------------- head layer 2: bn1+relu fused into A, [B,128] x [128,64] ----------------
__global__ __launch_bounds__(256) void head2_kernel(
    const float* __restrict__ u1,    // [B,128]
    const float* __restrict__ bn1,   // [256]: scale | shift
    const float* __restrict__ w2,    // [64,128]
    const float* __restrict__ b2,    // [64]
    float* __restrict__ u2,          // [B,64]
    float* __restrict__ stat2)       // [128]: sum | sumsq
{
  __shared__ _Float16 sW[64 * 128];
  __shared__ float sScale[128], sShift[128];
  const int tid = threadIdx.x;
  for (int i = tid; i < 64 * 128; i += 256) sW[i] = (_Float16)w2[i];
  if (tid < 128) { sScale[tid] = bn1[tid]; sShift[tid] = bn1[128 + tid]; }
  __syncthreads();

  const int wave = tid >> 5, lane = tid & 31, lm = lane & 15, hi = lane >> 4;
  const int r0 = (blockIdx.x * 8 + wave) * 16;
  const v8f zero8 = {0.f, 0.f, 0.f, 0.f, 0.f, 0.f, 0.f, 0.f};

  v16h af[4];
#pragma unroll
  for (int kc = 0; kc < 4; ++kc) {
#pragma unroll
    for (int vg = 0; vg < 8; ++vg) {
      const int k = kc * 32 + kA(vg, hi);
      float v0 = u1[(size_t)(r0 + lm) * 128 + k]     * sScale[k]     + sShift[k];
      float v1 = u1[(size_t)(r0 + lm) * 128 + k + 1] * sScale[k + 1] + sShift[k + 1];
      v0 = v0 > 0.f ? v0 : 0.f;
      v1 = v1 > 0.f ? v1 : 0.f;
      af[kc][2 * vg]     = (_Float16)v0;
      af[kc][2 * vg + 1] = (_Float16)v1;
    }
  }
#pragma unroll
  for (int nt = 0; nt < 4; ++nt) {
    v8f d = zero8;
#pragma unroll
    for (int kc = 0; kc < 4; ++kc) {
      v16h bf;
#pragma unroll
      for (int vg = 0; vg < 8; ++vg) {
        const int k = kc * 32 + kB(vg, hi);
        bf[2 * vg]     = sW[(nt * 16 + lm) * 128 + k];
        bf[2 * vg + 1] = sW[(nt * 16 + lm) * 128 + k + 1];
      }
      d = WMMA_F16(af[kc], bf, d);
    }
    const int c = nt * 16 + lm;
    const float bias = b2[c];
    float lsum = 0.f, lsq = 0.f;
#pragma unroll
    for (int j = 0; j < 8; ++j) {
      float v = d[j] + bias;
      u2[(size_t)(r0 + 8 * hi + j) * 64 + c] = v;
      lsum += v; lsq += v * v;
    }
    atomicAdd(&stat2[c], lsum);
    atomicAdd(&stat2[64 + c], lsq);
  }
}

// ---------------- head layer 3: bn2+relu, 64-dot, sigmoid ----------------
__global__ void head3_kernel(const float* __restrict__ u2,
                             const float* __restrict__ bn2,
                             const float* __restrict__ w3,
                             const float* __restrict__ b3,
                             float* __restrict__ out) {
  int b = blockIdx.x * blockDim.x + threadIdx.x;
  if (b >= B_) return;
  float s = 0.f;
#pragma unroll 8
  for (int c = 0; c < 64; ++c) {
    float v = u2[(size_t)b * 64 + c] * bn2[c] + bn2[64 + c];
    v = v > 0.f ? v : 0.f;
    s += v * w3[c];
  }
  s += b3[0];
  out[b] = 1.f / (1.f + __expf(-s));
}

// ---------------- launcher ----------------
extern "C" void kernel_launch(void* const* d_in, const int* in_sizes, int n_in,
                              void* d_out, int out_size, void* d_ws, size_t ws_size,
                              hipStream_t stream) {
  (void)in_sizes; (void)n_in; (void)out_size; (void)ws_size;
  const float* x        = (const float*)d_in[0];
  const float* x_scalar = (const float*)d_in[1];
  const int*   lengths  = (const int*)d_in[2];   // sample_indices row 0
  const float* phi_w1   = (const float*)d_in[3];
  const float* phi_b1   = (const float*)d_in[4];
  const float* phi_w2   = (const float*)d_in[5];
  const float* phi_b2   = (const float*)d_in[6];
  const float* mlp_w1   = (const float*)d_in[7];
  const float* mlp_b1   = (const float*)d_in[8];
  const float* bn1_g    = (const float*)d_in[9];
  const float* bn1_b    = (const float*)d_in[10];
  const float* mlp_w2   = (const float*)d_in[11];
  const float* mlp_b2   = (const float*)d_in[12];
  const float* bn2_g    = (const float*)d_in[13];
  const float* bn2_b    = (const float*)d_in[14];
  const float* mlp_w3   = (const float*)d_in[15];
  const float* mlp_b3   = (const float*)d_in[16];
  float* out = (float*)d_out;

  // workspace carving (256B aligned)
  char* p = (char*)d_ws;
  auto carve = [&](size_t bytes) -> char* {
    char* r = p;
    p += (bytes + 255) & ~(size_t)255;
    return r;
  };
  int*      offsets = (int*)carve((B_ + 1) * sizeof(int));
  int*      seg     = (int*)carve((size_t)T_ * sizeof(int));
  float*    pooled  = (float*)carve((size_t)B_ * 64 * sizeof(float));
  _Float16* zh      = (_Float16*)carve((size_t)B_ * 96 * sizeof(_Float16));
  float*    u1      = (float*)carve((size_t)B_ * 128 * sizeof(float));
  float*    u2      = (float*)carve((size_t)B_ * 64 * sizeof(float));
  float*    stat1   = (float*)carve(256 * sizeof(float));
  float*    bn1     = (float*)carve(256 * sizeof(float));
  float*    stat2   = (float*)carve(128 * sizeof(float));
  float*    bn2     = (float*)carve(128 * sizeof(float));

  // re-zero accumulators every call (deterministic / graph-replay safe)
  zero_kernel<<<(B_ * 64 + 255) / 256, 256, 0, stream>>>(pooled, B_ * 64);
  zero_kernel<<<1, 256, 0, stream>>>(stat1, 256);
  zero_kernel<<<1, 128, 0, stream>>>(stat2, 128);

  scan_offsets<<<1, 256, 0, stream>>>(lengths, offsets);
  seg_fill<<<B_ / 256, 256, 0, stream>>>(offsets, seg);

  phi_pool_kernel<<<T_ / (8 * TPW * 16), 256, 0, stream>>>(
      x, phi_w1, phi_b1, phi_w2, phi_b2, seg, pooled);

  zfin_kernel<<<(B_ * 96 + 255) / 256, 256, 0, stream>>>(pooled, x_scalar, lengths, zh);

  head1_kernel<<<B_ / (16 * 8), 256, 0, stream>>>(zh, mlp_w1, mlp_b1, u1, stat1);
  bnfin_kernel<<<1, 128, 0, stream>>>(stat1, bn1_g, bn1_b, bn1, 128, (float)B_);
  head2_kernel<<<B_ / (16 * 8), 256, 0, stream>>>(u1, bn1, mlp_w2, mlp_b2, u2, stat2);
  bnfin_kernel<<<1, 64, 0, stream>>>(stat2, bn2_g, bn2_b, bn2, 64, (float)B_);
  head3_kernel<<<B_ / 256, 256, 0, stream>>>(u2, bn2, mlp_w3, mlp_b3, out);
}
